// CustomMHA_63780264346209
// MI455X (gfx1250) — compile-verified, matrix operand order checked by reading, not run on hardware
//
#include <hip/hip_runtime.h>
#include <hip/hip_bf16.h>

// ---------------------------------------------------------------------------
// MHA forward for MI455X (gfx1250, wave32, WMMA).
//   x:(4,2048,1024) f32, W_qkv:(3072,1024) f32, W_o:(1024,1024) f32
//   Y:(4,2048,1024) f32
// Pipeline: f32->bf16 convert, QKV GEMM (WMMA bf16), flash attention
// (WMMA bf16, online softmax with ds_swizzle butterflies + WMMA row-sum),
// output GEMM (-> f32).
// ---------------------------------------------------------------------------

typedef __bf16 bf16_t;
typedef __attribute__((ext_vector_type(4)))  __bf16 v4bf;
typedef __attribute__((ext_vector_type(8)))  __bf16 v8bf;
typedef __attribute__((ext_vector_type(16))) __bf16 v16bf;
typedef __attribute__((ext_vector_type(8)))  float  v8f;

constexpr int D_MODEL = 1024;
constexpr int N_HEADS = 16;
constexpr int BB      = 4;
constexpr int SS      = 2048;
constexpr int DH      = 64;       // head dim
constexpr int TOKENS  = BB * SS;  // 8192
constexpr int CHUNK   = 64;       // keys per flash iteration

#define WMMA_BF16(A, B, C) \
    __builtin_amdgcn_wmma_f32_16x16x32_bf16(false, (A), false, (B), (short)0, (C), false, false)

static __device__ inline v16bf cat16(v8bf lo, v8bf hi) {
    return __builtin_shufflevector(lo, hi, 0, 1, 2, 3, 4, 5, 6, 7,
                                           8, 9, 10, 11, 12, 13, 14, 15);
}

// Load a 16x32 bf16 WMMA A/B fragment whose lane data are the two contiguous
// 8-element K-runs at (row, k0 + 8*half) and (row, k0 + 16 + 8*half).
static __device__ inline v16bf load_frag_g(const bf16_t* __restrict__ base) {
    v8bf lo = *reinterpret_cast<const v8bf*>(base);
    v8bf hi = *reinterpret_cast<const v8bf*>(base + 16);
    return cat16(lo, hi);
}

// One batched butterfly step of a max-reduction over 16-lane groups using
// ds_swizzle_b32 group-of-32 xor mode: offset = (xor_mask<<10) | 0x1F.
// All 8 rows issue their swizzle before the single dscnt wait.
template <int IMM>
static __device__ inline void swz_max_step(float (&vm)[8]) {
    float t[8];
#pragma unroll
    for (int v = 0; v < 8; ++v)
        t[v] = __int_as_float(
            __builtin_amdgcn_ds_swizzle(__float_as_int(vm[v]), IMM));
#pragma unroll
    for (int v = 0; v < 8; ++v) vm[v] = fmaxf(vm[v], t[v]);
}

// ---------------------------------------------------------------------------
// Kernel 0: f32 -> bf16 conversion, 4 elements per thread.
// ---------------------------------------------------------------------------
__global__ void __launch_bounds__(256)
cvt_f32_bf16(const float* __restrict__ in, bf16_t* __restrict__ out, int n) {
    int i = (blockIdx.x * blockDim.x + threadIdx.x) * 4;
    if (i < n) {
        float4 f = *reinterpret_cast<const float4*>(in + i);
        v4bf o;
        o[0] = (bf16_t)f.x; o[1] = (bf16_t)f.y;
        o[2] = (bf16_t)f.z; o[3] = (bf16_t)f.w;
        *reinterpret_cast<v4bf*>(out + i) = o;
    }
}

// ---------------------------------------------------------------------------
// GEMM: C(M,N) = A(M,K) * Bw(N,K)^T, bf16 inputs, f32 accumulate.
// Wave tile 64x64 (4x4 fragments). Block = 8 waves covering 8 N-tiles.
// MODE 0: scatter into Q (scaled by 0.125), K, V as (B,H,S,DH) bf16.
// MODE 1: store f32 rows straight to Y.
// ---------------------------------------------------------------------------
template <int MODE>
__global__ void __launch_bounds__(256)
gemm_bf16(const bf16_t* __restrict__ A, const bf16_t* __restrict__ Bw,
          int N, int K,
          bf16_t* __restrict__ Qb, bf16_t* __restrict__ Kb,
          bf16_t* __restrict__ Vb, float* __restrict__ Yout) {
    const int wave = threadIdx.x >> 5;
    const int lane = threadIdx.x & 31;
    const int r    = lane & 15;
    const int half = lane >> 4;

    const int m0 = blockIdx.x * 64;
    const int n0 = (blockIdx.y * 8 + wave) * 64;
    if (n0 >= N) return;

    v8f zero = {};
    v8f acc[4][4];
#pragma unroll
    for (int i = 0; i < 4; ++i)
#pragma unroll
        for (int j = 0; j < 4; ++j) acc[i][j] = zero;

    for (int k0 = 0; k0 < K; k0 += 32) {
        const int ko = k0 + 8 * half;
        v16bf aF[4], bF[4];
#pragma unroll
        for (int i = 0; i < 4; ++i)
            aF[i] = load_frag_g(A + (size_t)(m0 + 16 * i + r) * K + ko);
#pragma unroll
        for (int j = 0; j < 4; ++j)
            bF[j] = load_frag_g(Bw + (size_t)(n0 + 16 * j + r) * K + ko);
#pragma unroll
        for (int i = 0; i < 4; ++i)
#pragma unroll
            for (int j = 0; j < 4; ++j)
                acc[i][j] = WMMA_BF16(aF[i], bF[j], acc[i][j]);
    }

    // Epilogue. C layout: lane holds column n0+16j+r; VGPR v holds row
    // m0+16i+v+8*half.
#pragma unroll
    for (int j = 0; j < 4; ++j) {
        const int e = n0 + 16 * j + r;  // output column
        if (MODE == 0) {
            const int sel = e >> 10;           // 0:Q 1:K 2:V
            const int rem = e & 1023;
            const int h   = rem >> 6;
            const int d   = rem & 63;
            bf16_t* dst = (sel == 0) ? Qb : (sel == 1) ? Kb : Vb;
            const float scale = (sel == 0) ? 0.125f : 1.0f;  // 1/sqrt(64) on Q
#pragma unroll
            for (int i = 0; i < 4; ++i) {
#pragma unroll
                for (int v = 0; v < 8; ++v) {
                    const int tok = m0 + 16 * i + v + 8 * half;
                    const int b = tok >> 11;       // / SS
                    const int s = tok & (SS - 1);
                    const size_t idx =
                        (((size_t)(b * N_HEADS + h)) * SS + s) * DH + d;
                    dst[idx] = (bf16_t)(acc[i][j][v] * scale);
                }
            }
        } else {
#pragma unroll
            for (int i = 0; i < 4; ++i) {
#pragma unroll
                for (int v = 0; v < 8; ++v) {
                    const int tok = m0 + 16 * i + v + 8 * half;
                    Yout[(size_t)tok * D_MODEL + e] = acc[i][j][v];
                }
            }
        }
    }
}

// ---------------------------------------------------------------------------
// Flash attention. Grid: (S/64, B*H). Block: 128 threads (4 waves).
// Each wave owns a 16-query tile; 64 keys per iteration (18 WMMAs/chunk).
// Row-sum of exp() is accumulated with a WMMA against an all-ones B matrix;
// row-max uses 4 batched ds_swizzle butterfly steps. Q pre-scaled by 1/8.
// ---------------------------------------------------------------------------
__global__ void __launch_bounds__(128)
flash_attn(const bf16_t* __restrict__ Qb, const bf16_t* __restrict__ Kb,
           const bf16_t* __restrict__ Vb, bf16_t* __restrict__ Ob) {
    __shared__ bf16_t VtS[DH * CHUNK];      // V^T chunk: row d (128B), 64 keys
    __shared__ bf16_t PS[4][16 * CHUNK];    // per-wave P tile (16 q x 64 k)

    const int wave = threadIdx.x >> 5;
    const int lane = threadIdx.x & 31;
    const int r    = lane & 15;
    const int half = lane >> 4;

    const int bh    = blockIdx.y;                 // b*N_HEADS + h
    const int qbase = blockIdx.x * 64 + wave * 16;
    const size_t base = (size_t)bh * SS * DH;     // into (B,H,S,DH)

    // Q A-fragments (16 x 64 split into two 16x32 fragments), loaded once.
    v16bf qF[2];
#pragma unroll
    for (int g = 0; g < 2; ++g)
        qF[g] = load_frag_g(Qb + base + (size_t)(qbase + r) * DH + 32 * g + 8 * half);

    // All-ones B fragment for the WMMA row-sum trick.
    v16bf ones;
#pragma unroll
    for (int t = 0; t < 16; ++t) ones[t] = (bf16_t)1.0f;

    v8f zero = {};
    v8f oAcc[4];
#pragma unroll
    for (int t = 0; t < 4; ++t) oAcc[t] = zero;
    v8f lsum = zero;
    float mrow[8];
#pragma unroll
    for (int v = 0; v < 8; ++v) mrow[v] = -1e30f;

    for (int kb = 0; kb < SS; kb += CHUNK) {
        __syncthreads();  // protect VtS reuse
        // Cooperative stage of V^T chunk: thread -> (key s_l, 32-dim group).
        {
            const int s_l = threadIdx.x >> 1;         // 0..63
            const int dg  = (threadIdx.x & 1) * 32;   // 0 or 32
            const bf16_t* vp = Vb + base + (size_t)(kb + s_l) * DH + dg;
            v8bf v0 = *reinterpret_cast<const v8bf*>(vp);
            v8bf v1 = *reinterpret_cast<const v8bf*>(vp + 8);
            v8bf v2 = *reinterpret_cast<const v8bf*>(vp + 16);
            v8bf v3 = *reinterpret_cast<const v8bf*>(vp + 24);
#pragma unroll
            for (int tt = 0; tt < 8; ++tt) {
                VtS[(dg + tt) * CHUNK + s_l]      = v0[tt];
                VtS[(dg + 8 + tt) * CHUNK + s_l]  = v1[tt];
                VtS[(dg + 16 + tt) * CHUNK + s_l] = v2[tt];
                VtS[(dg + 24 + tt) * CHUNK + s_l] = v3[tt];
            }
        }
        __syncthreads();

        // Scores: S = Q * K^T for four 16x16 key tiles.
        v8f sF[4];
#pragma unroll
        for (int j = 0; j < 4; ++j) {
            const int key = kb + 16 * j + r;
            const bf16_t* kp = Kb + base + (size_t)key * DH + 8 * half;
            v8f s = zero;
            s = WMMA_BF16(qF[0], load_frag_g(kp), s);
            s = WMMA_BF16(qF[1], load_frag_g(kp + 32), s);
            sF[j] = s;
        }

        // Row max over 64 keys: local max of 4 tiles, then 4 batched
        // butterfly steps across the 16 lanes of each half-wave.
        float vm[8];
#pragma unroll
        for (int v = 0; v < 8; ++v)
            vm[v] = fmaxf(fmaxf(sF[0][v], sF[1][v]),
                          fmaxf(sF[2][v], sF[3][v]));
        swz_max_step<0x041F>(vm);   // lane ^ 1
        swz_max_step<0x081F>(vm);   // lane ^ 2
        swz_max_step<0x101F>(vm);   // lane ^ 4
        swz_max_step<0x201F>(vm);   // lane ^ 8

        // Online rescale + exponentiation (sum comes from WMMA below).
#pragma unroll
        for (int v = 0; v < 8; ++v) {
            const float mn    = fmaxf(mrow[v], vm[v]);
            const float alpha = __expf(mrow[v] - mn);
            mrow[v] = mn;
#pragma unroll
            for (int j = 0; j < 4; ++j) sF[j][v] = __expf(sF[j][v] - mn);
            lsum[v] *= alpha;
#pragma unroll
            for (int t = 0; t < 4; ++t) oAcc[t][v] *= alpha;
        }

        // C-layout -> A-layout transpose of P through LDS (bf16).
        bf16_t* P = &PS[wave][0];
#pragma unroll
        for (int j = 0; j < 4; ++j)
#pragma unroll
            for (int v = 0; v < 8; ++v)
                P[(v + 8 * half) * CHUNK + 16 * j + r] = (bf16_t)sF[j][v];

        v16bf pF[2];
#pragma unroll
        for (int g = 0; g < 2; ++g) {
            const bf16_t* pp = P + r * CHUNK + 32 * g + 8 * half;
            pF[g] = cat16(*reinterpret_cast<const v8bf*>(pp),
                          *reinterpret_cast<const v8bf*>(pp + 16));
        }

        // Row sums: lsum += P * ones (every lane gets its rows' sums).
        lsum = WMMA_BF16(pF[0], ones, lsum);
        lsum = WMMA_BF16(pF[1], ones, lsum);

        // O += P * V   (V B-fragments contiguous from transposed LDS tile)
#pragma unroll
        for (int t = 0; t < 4; ++t) {
#pragma unroll
            for (int g = 0; g < 2; ++g) {
                const bf16_t* vtp = VtS + (16 * t + r) * CHUNK + 32 * g + 8 * half;
                v16bf vF = cat16(*reinterpret_cast<const v8bf*>(vtp),
                                 *reinterpret_cast<const v8bf*>(vtp + 16));
                oAcc[t] = WMMA_BF16(pF[g], vF, oAcc[t]);
            }
        }
    }

    // Normalize and store O into (B,S,D) bf16 (heads re-merged).
    const int b = bh >> 4;
    const int h = bh & 15;
#pragma unroll
    for (int v = 0; v < 8; ++v) {
        const float inv = 1.0f / lsum[v];
#pragma unroll
        for (int t = 0; t < 4; ++t) oAcc[t][v] *= inv;
    }
#pragma unroll
    for (int t = 0; t < 4; ++t) {
        const int col = h * DH + 16 * t + r;
#pragma unroll
        for (int v = 0; v < 8; ++v) {
            const int q = qbase + v + 8 * half;
            Ob[((size_t)(b * SS + q)) * D_MODEL + col] = (bf16_t)oAcc[t][v];
        }
    }
}

// ---------------------------------------------------------------------------
// Host-side launch
// ---------------------------------------------------------------------------
extern "C" void kernel_launch(void* const* d_in, const int* in_sizes, int n_in,
                              void* d_out, int out_size, void* d_ws, size_t ws_size,
                              hipStream_t stream) {
    (void)in_sizes; (void)n_in; (void)out_size; (void)ws_size;
    const float* x     = (const float*)d_in[0];
    const float* W_qkv = (const float*)d_in[1];
    const float* W_o   = (const float*)d_in[2];
    float* Y = (float*)d_out;

    // Workspace layout (bf16 buffers), ~88 MB total.
    char* w = (char*)d_ws;
    size_t off = 0;
    bf16_t* xb    = (bf16_t*)(w + off); off += (size_t)TOKENS * D_MODEL * 2;      // 16 MB
    bf16_t* wqkvb = (bf16_t*)(w + off); off += (size_t)3 * D_MODEL * D_MODEL * 2; //  6 MB
    bf16_t* wob   = (bf16_t*)(w + off); off += (size_t)D_MODEL * D_MODEL * 2;     //  2 MB
    bf16_t* Qb    = (bf16_t*)(w + off); off += (size_t)TOKENS * D_MODEL * 2;      // 16 MB
    bf16_t* Kb    = (bf16_t*)(w + off); off += (size_t)TOKENS * D_MODEL * 2;      // 16 MB
    bf16_t* Vb    = (bf16_t*)(w + off); off += (size_t)TOKENS * D_MODEL * 2;      // 16 MB
    bf16_t* Ob    = (bf16_t*)(w + off); off += (size_t)TOKENS * D_MODEL * 2;      // 16 MB

    const int nx    = TOKENS * D_MODEL;
    const int nwqkv = 3 * D_MODEL * D_MODEL;
    const int nwo   = D_MODEL * D_MODEL;
    cvt_f32_bf16<<<(nx / 4 + 255) / 256, 256, 0, stream>>>(x, xb, nx);
    cvt_f32_bf16<<<(nwqkv / 4 + 255) / 256, 256, 0, stream>>>(W_qkv, wqkvb, nwqkv);
    cvt_f32_bf16<<<(nwo / 4 + 255) / 256, 256, 0, stream>>>(W_o, wob, nwo);

    // QKV GEMM: M=8192, N=3072, K=1024. 128 M-tiles x 48 N-tiles.
    gemm_bf16<0><<<dim3(TOKENS / 64, 6), 256, 0, stream>>>(
        xb, wqkvb, 3 * D_MODEL, D_MODEL, Qb, Kb, Vb, nullptr);

    // Attention: 32 q-tiles of 64 per (b,h), 64 (b,h) pairs.
    flash_attn<<<dim3(SS / 64, BB * N_HEADS), 128, 0, stream>>>(Qb, Kb, Vb, Ob);

    // Output GEMM: M=8192, N=1024, K=1024 -> f32 Y.
    gemm_bf16<1><<<dim3(TOKENS / 64, 2), 256, 0, stream>>>(
        Ob, wob, D_MODEL, D_MODEL, nullptr, nullptr, nullptr, Y);
}
